// LocalAstrometryMatcher_41532333752737
// MI455X (gfx1250) — matrix operand order repaired and checked by reading, not compiled, not run on hardware
//
#include <hip/hip_runtime.h>
#include <math.h>

typedef __attribute__((ext_vector_type(16))) _Float16 v16h;
typedef __attribute__((ext_vector_type(8)))  _Float16 v8h;
typedef __attribute__((ext_vector_type(8)))  float    v8f;

// ---------------------------------------------------------------- helpers

static __device__ inline float geluf(float x) {
  return 0.5f * x * (1.0f + erff(x * 0.70710678118654752440f));
}

// A-fragment (16x32 f16, M=lane%16): lane<16 holds K=0..7,16..23 ; lane>=16 K=8..15,24..31.
// p points at 32 contiguous f16 K-values for this row. Two 16B chunks: {0,2} or {1,3}.
static __device__ inline v16h fragA16(const _Float16* p, bool hi) {
  const v8h* q = (const v8h*)p;
  v8h lo = q[hi ? 1 : 0];
  v8h h8 = q[hi ? 3 : 2];
  return __builtin_shufflevector(lo, h8, 0,1,2,3,4,5,6,7,8,9,10,11,12,13,14,15);
}

// B-fragment (32x16 f16, N=lane%16): lane<16 holds K=0..7,16..23 ; lane>=16 K=8..15,24..31.
// p points at 32 contiguous f16 K-values for this column. Chunks {0,2} or {1,3}.
static __device__ inline v16h fragB16(const _Float16* p, bool hi) {
  const v8h* q = (const v8h*)p;
  v8h lo = q[hi ? 1 : 0];
  v8h h8 = q[hi ? 3 : 2];
  return __builtin_shufflevector(lo, h8, 0,1,2,3,4,5,6,7,8,9,10,11,12,13,14,15);
}

static __device__ inline float gaussUn(int p) {
  int y = p >> 6, x = p & 63;
  float fy = -1.0f + (float)y * (2.0f / 63.0f);
  float fx = -1.0f + (float)x * (2.0f / 63.0f);
  return expf(-2.0f * (fy * fy + fx * fx));   // exp(-r^2 / (2*0.5^2))
}

// ---------------------------------------------------------------- pack NCHW f32 -> NHWC32 f16

__global__ __launch_bounds__(256) void pack_nhwc32(const float* __restrict__ src,
                                                   _Float16* __restrict__ dst, int Cin) {
  size_t i = (size_t)blockIdx.x * 256 + threadIdx.x;   // 256*4096*32 elements
  int c = (int)(i & 31);
  size_t bp = i >> 5;                 // b*4096 + p
  size_t b = bp >> 12, p = bp & 4095;
  dst[i] = (c < Cin) ? (_Float16)src[(b * Cin + c) * 4096 + p] : (_Float16)0.0f;
}

// ---------------------------------------------------------------- fused conv3x3 + GroupNorm(4) + GELU
// One block per image. Implicit GEMM: M=4096 spatial, N=32 out-ch, K=9 taps x 32 ch.
// Each wave owns an m-tile and accumulates BOTH n-tiles from one A load (A reuse 2x).

__global__ __launch_bounds__(256) void conv_gn_gelu(const _Float16* __restrict__ in,
                                                    _Float16* __restrict__ out,
                                                    const float* __restrict__ W, int Cin,
                                                    const float* __restrict__ gamma,
                                                    const float* __restrict__ beta) {
  __shared__ __align__(16) _Float16 wl[9 * 32 * 32];
  __shared__ float sSum[4], sSum2[4];
  __shared__ float sGam[32], sBet[32];

  const int tid = threadIdx.x;
  const int img = blockIdx.x;
  in  += (size_t)img * 131072;
  out += (size_t)img * 131072;

  for (int i = tid; i < 9 * 1024; i += 256) {   // stage weights [t][co][ci], zero-pad ci>=Cin
    int t = i >> 10, rem = i & 1023, co = rem >> 5, ci = rem & 31;
    float v = (ci < Cin) ? W[(co * Cin + ci) * 9 + t] : 0.0f;
    wl[i] = (_Float16)v;
  }
  if (tid < 4)  { sSum[tid] = 0.0f; sSum2[tid] = 0.0f; }
  if (tid < 32) { sGam[tid] = gamma[tid]; sBet[tid] = beta[tid]; }
  __syncthreads();

  const int wave = tid >> 5;
  const int lane = tid & 31;
  const bool hi = lane >= 16;
  const int m = lane & 15;
  const int co0 = m, co1 = 16 + m;
  const int g0 = co0 >> 3, g1 = co1 >> 3;

  for (int mtile = wave; mtile < 256; mtile += 8) {     // 32 uniform iterations per wave
    int m0 = mtile << 4;
    int y = m0 >> 6, x0 = m0 & 63;
    v8f acc0 = {}, acc1 = {};
    // prefetch this wave's next m-tile pixel row into caches (gfx1250 global_prefetch)
    __builtin_prefetch((const void*)(in + ((size_t)((m0 + 128) & 4095) << 5)), 0, 3);
    #pragma unroll
    for (int t = 0; t < 9; ++t) {
      int dy = t / 3 - 1, dx = t % 3 - 1;
      int yy = y + dy, xx = x0 + m + dx;
      v16h a;
      if ((unsigned)yy < 64u && (unsigned)xx < 64u) {
        a = fragA16(in + ((size_t)(yy * 64 + xx) << 5), hi);
      } else { v16h z = {}; a = z; }                     // SAME zero padding
      v16h b0 = fragB16(&wl[(t << 10) + (co0 << 5)], hi);
      v16h b1 = fragB16(&wl[(t << 10) + (co1 << 5)], hi);
      acc0 = __builtin_amdgcn_wmma_f32_16x16x32_f16(false, a, false, b0, (short)0, acc0, false, false);
      acc1 = __builtin_amdgcn_wmma_f32_16x16x32_f16(false, a, false, b1, (short)0, acc1, false, false);
    }
    float s0 = 0.0f, s20 = 0.0f, s1 = 0.0f, s21 = 0.0f;
    #pragma unroll
    for (int r = 0; r < 8; ++r) {
      int pos = m0 + r + (hi ? 8 : 0);
      float v0 = acc0[r], v1 = acc1[r];
      out[((size_t)pos << 5) + co0] = (_Float16)v0;      // raw conv output (f16)
      out[((size_t)pos << 5) + co1] = (_Float16)v1;
      s0 += v0; s20 += v0 * v0;
      s1 += v1; s21 += v1 * v1;
    }
    atomicAdd(&sSum[g0], s0);  atomicAdd(&sSum2[g0], s20);
    atomicAdd(&sSum[g1], s1);  atomicAdd(&sSum2[g1], s21);
  }
  __threadfence_block();
  __syncthreads();

  float mean[4], rsig[4];
  #pragma unroll
  for (int g = 0; g < 4; ++g) {
    float mu = sSum[g] * (1.0f / 32768.0f);
    float var = sSum2[g] * (1.0f / 32768.0f) - mu * mu;
    mean[g] = mu; rsig[g] = rsqrtf(var + 1e-5f);
  }
  // vectorized in-place GN + GELU: each 8-half chunk is exactly one GN group
  v8h* q = (v8h*)out;
  for (int e8 = tid; e8 < 16384; e8 += 256) {
    int g = e8 & 3;                 // chunk-in-pixel == group (8 ch per group)
    int cb = g << 3;
    v8h v = q[e8];
    v8h r;
    #pragma unroll
    for (int j = 0; j < 8; ++j) {
      float x = (float)v[j];
      float xn = (x - mean[g]) * rsig[g] * sGam[cb + j] + sBet[cb + j];
      r[j] = (_Float16)geluf(xn);
    }
    q[e8] = r;
  }
}

// ---------------------------------------------------------------- fused 1x1 proj + L2-norm (+sw weighting) + gauss pooling

__global__ __launch_bounds__(256) void proj_l2n(const _Float16* __restrict__ feat,
                                                _Float16* __restrict__ out,
                                                const float* __restrict__ P,
                                                float* __restrict__ poolOut,
                                                float* __restrict__ swBuf, int doSw) {
  __shared__ __align__(16) _Float16 pw[1024];
  __shared__ float sGsum, sSw, sPool[32];

  const int tid = threadIdx.x;
  const int img = blockIdx.x;
  feat  += (size_t)img * 131072;
  out   += (size_t)img * 131072;
  swBuf += (size_t)img * 4096;

  if (tid == 0) { sGsum = 0.0f; sSw = 0.0f; }
  if (tid < 32) sPool[tid] = 0.0f;
  __syncthreads();
  for (int i = tid; i < 1024; i += 256) pw[i] = (_Float16)P[i];   // [co][ci]
  {
    float gl = 0.0f;
    for (int p = tid; p < 4096; p += 256) gl += gaussUn(p);
    atomicAdd(&sGsum, gl);
  }
  __syncthreads();

  const int wave = tid >> 5, lane = tid & 31;
  const bool hi = lane >= 16;
  const int m = lane & 15;

  for (int mtile = wave; mtile < 256; mtile += 8) {  // K=32 -> one A load, two WMMAs
    int m0 = mtile << 4;
    v16h a  = fragA16(feat + ((size_t)(m0 + m) << 5), hi);
    v16h b0 = fragB16(&pw[m << 5], hi);
    v16h b1 = fragB16(&pw[(16 + m) << 5], hi);
    v8f acc0 = {}, acc1 = {};
    acc0 = __builtin_amdgcn_wmma_f32_16x16x32_f16(false, a, false, b0, (short)0, acc0, false, false);
    acc1 = __builtin_amdgcn_wmma_f32_16x16x32_f16(false, a, false, b1, (short)0, acc1, false, false);
    #pragma unroll
    for (int r = 0; r < 8; ++r) {
      int pos = m0 + r + (hi ? 8 : 0);
      out[((size_t)pos << 5) + m]      = (_Float16)acc0[r];
      out[((size_t)pos << 5) + 16 + m] = (_Float16)acc1[r];
    }
  }
  __threadfence_block();
  __syncthreads();

  {
    float pl[32];
    #pragma unroll
    for (int c = 0; c < 32; ++c) pl[c] = 0.0f;
    float swl = 0.0f;
    float invG = 1.0f / sGsum;
    for (int p = tid; p < 4096; p += 256) {
      _Float16* px = out + ((size_t)p << 5);
      float ss = 0.0f;
      #pragma unroll
      for (int c = 0; c < 32; ++c) { float v = (float)px[c]; ss += v * v; }
      float nf = 1.0f / fmaxf(sqrtf(ss), 1e-6f);
      #pragma unroll
      for (int c = 0; c < 32; ++c) px[c] = (_Float16)((float)px[c] * nf);  // L2 normalize in place
      float gn = gaussUn(p) * invG;
      const _Float16* fp = feat + ((size_t)p << 5);
      #pragma unroll
      for (int c = 0; c < 32; ++c) pl[c] += (float)fp[c] * gn;             // gauss pooling of features
      if (doSw) {
        float energy = ss * nf * nf;     // ==1 unless degenerate norm, matches reference
        float sw = energy * gn;
        swBuf[p] = sw;
        swl += sw;
      }
    }
    #pragma unroll
    for (int c = 0; c < 32; ++c) atomicAdd(&sPool[c], pl[c]);
    if (doSw) atomicAdd(&sSw, swl);
  }
  __threadfence_block();
  __syncthreads();

  if (doSw) {
    float inv = 1.0f / (sSw + 1e-8f);
    v8h* q = (v8h*)out;
    for (int e8 = tid; e8 < 16384; e8 += 256) {        // rw = n * normalized sw (vectorized)
      int p = e8 >> 2;
      float sc = swBuf[p] * inv;
      v8h v = q[e8];
      v8h r;
      #pragma unroll
      for (int j = 0; j < 8; ++j) r[j] = (_Float16)((float)v[j] * sc);
      q[e8] = r;
    }
  }
  if (tid < 32) poolOut[(size_t)img * 32 + tid] = sPool[tid];
}

// ---------------------------------------------------------------- 7x7 shift correlation logits

__global__ __launch_bounds__(256) void corr_logits(const _Float16* __restrict__ rw,
                                                   const _Float16* __restrict__ vn,
                                                   float* __restrict__ logBuf,
                                                   float* __restrict__ outp) {
  __shared__ float red[49];
  const int tid = threadIdx.x;
  const int img = blockIdx.x;
  rw += (size_t)img * 131072;
  vn += (size_t)img * 131072;
  if (tid < 49) red[tid] = 0.0f;
  __syncthreads();

  float lg[49];
  #pragma unroll
  for (int s = 0; s < 49; ++s) lg[s] = 0.0f;

  for (int p = tid; p < 4096; p += 256) {
    int y = p >> 6, x = p & 63;
    float rv[32];
    const _Float16* rp = rw + ((size_t)p << 5);
    #pragma unroll
    for (int c = 0; c < 32; ++c) rv[c] = (float)rp[c];
    #pragma unroll
    for (int s = 0; s < 49; ++s) {
      int dy = s / 7 - 3, dx = s % 7 - 3;
      int yy = y + dy; yy = yy < 0 ? 0 : (yy > 63 ? 63 : yy);   // edge padding
      int xx = x + dx; xx = xx < 0 ? 0 : (xx > 63 ? 63 : xx);
      const _Float16* vp = vn + ((size_t)(yy * 64 + xx) << 5);
      float d = 0.0f;
      #pragma unroll
      for (int c = 0; c < 32; ++c) d += rv[c] * (float)vp[c];
      lg[s] += d;
    }
  }
  #pragma unroll
  for (int s = 0; s < 49; ++s) atomicAdd(&red[s], lg[s]);
  __syncthreads();
  if (tid < 49) {
    float v = red[tid] * 0.17677669529663687f;   // 1/sqrt(32)
    logBuf[(size_t)img * 49 + tid] = v;
    outp[(size_t)img * 54 + 5 + tid] = v;
  }
}

// ---------------------------------------------------------------- MLP head (WMMA) + softmax + finalize

static __device__ void gemm256x128x128(const _Float16* __restrict__ A,
                                       const _Float16* __restrict__ wl,
                                       const float* __restrict__ bias,
                                       _Float16* __restrict__ C, int act) {
  const int tid = threadIdx.x;
  const int wave = tid >> 5, lane = tid & 31;
  const bool hi = lane >= 16;
  const int m = lane & 15;
  for (int tile = wave; tile < 128; tile += 8) {
    int mtile = tile >> 3, nt = tile & 7;
    int m0 = mtile << 4;
    int row = m0 + m;
    int col = (nt << 4) + m;
    v8f acc = {};
    #pragma unroll
    for (int kc = 0; kc < 4; ++kc) {
      v16h a = fragA16(A + (size_t)row * 128 + kc * 32, hi);
      v16h b = fragB16(wl + (size_t)col * 128 + kc * 32, hi);
      acc = __builtin_amdgcn_wmma_f32_16x16x32_f16(false, a, false, b, (short)0, acc, false, false);
    }
    float bb = bias[col];
    #pragma unroll
    for (int r = 0; r < 8; ++r) {
      int orow = m0 + r + (hi ? 8 : 0);
      float v = acc[r] + bb;
      if (act) v = geluf(v);
      C[(size_t)orow * 128 + col] = (_Float16)v;
    }
  }
}

__global__ __launch_bounds__(256) void mlp_head(const float* __restrict__ logBuf,
                                                const float* __restrict__ rp,
                                                const float* __restrict__ vp,
                                                const float* __restrict__ logTemp,
                                                const float* __restrict__ W1, const float* __restrict__ b1,
                                                const float* __restrict__ W2, const float* __restrict__ b2,
                                                const float* __restrict__ W3, const float* __restrict__ b3,
                                                const float* __restrict__ p2s,
                                                _Float16* __restrict__ pooled,
                                                _Float16* __restrict__ h1,
                                                _Float16* __restrict__ h2,
                                                float* __restrict__ outp) {
  __shared__ __align__(16) _Float16 wl[128 * 128];
  __shared__ float scdx[256], scdy[256];
  __shared__ float sres[256 * 3];

  const int tid = threadIdx.x;

  // Phase A: softmax over 49 logits, coarse shift, build pooled feature row (K padded 98 -> 128).
  {
    int b = tid;
    float l[49], mx = -1e30f;
    #pragma unroll
    for (int s = 0; s < 49; ++s) { l[s] = logBuf[(size_t)b * 49 + s]; mx = fmaxf(mx, l[s]); }
    float temp = fmaxf(expf(logTemp[0]), 1e-3f);
    float sum = 0.0f, cdx = 0.0f, cdy = 0.0f;
    #pragma unroll
    for (int s = 0; s < 49; ++s) {
      float e = expf((l[s] - mx) / temp);
      sum += e;
      cdx += e * (float)(s % 7 - 3);
      cdy += e * (float)(s / 7 - 3);
    }
    cdx /= sum; cdy /= sum;
    scdx[b] = cdx; scdy[b] = cdy;
    const float* rpb = rp + (size_t)b * 32;
    const float* vpb = vp + (size_t)b * 32;
    _Float16* pr = pooled + (size_t)b * 128;
    #pragma unroll
    for (int j = 0; j < 32; ++j) {
      pr[j]      = (_Float16)rpb[j];
      pr[32 + j] = (_Float16)vpb[j];
      pr[64 + j] = (_Float16)(rpb[j] - vpb[j]);
    }
    pr[96] = (_Float16)cdx;
    pr[97] = (_Float16)cdy;
    #pragma unroll
    for (int j = 98; j < 128; ++j) pr[j] = (_Float16)0.0f;
  }
  for (int i = tid; i < 16384; i += 256) {       // stage W1 transposed [n][k], zero-pad k>=98
    int n = i >> 7, k = i & 127;
    wl[i] = (k < 98) ? (_Float16)W1[(size_t)k * 128 + n] : (_Float16)0.0f;
  }
  __threadfence_block();
  __syncthreads();

  gemm256x128x128(pooled, wl, b1, h1, 1);
  __threadfence_block();
  __syncthreads();
  for (int i = tid; i < 16384; i += 256) {       // stage W2 transposed
    int n = i >> 7, k = i & 127;
    wl[i] = (_Float16)W2[(size_t)k * 128 + n];
  }
  __syncthreads();

  gemm256x128x128(h1, wl, b2, h2, 1);
  __threadfence_block();
  __syncthreads();
  for (int i = tid; i < 2048; i += 256) {        // stage W3 (128x3) transposed, N padded to 16
    int n = i >> 7, k = i & 127;
    wl[i] = (n < 3) ? (_Float16)W3[(size_t)k * 3 + n] : (_Float16)0.0f;
  }
  __syncthreads();

  {                                              // GEMM3: M=256, N=16(3 used), K=128
    const int wave = tid >> 5, lane = tid & 31;
    const bool hi = lane >= 16;
    const int m = lane & 15;
    for (int tile = wave; tile < 16; tile += 8) {
      int m0 = tile << 4;
      int row = m0 + m;
      v8f acc = {};
      #pragma unroll
      for (int kc = 0; kc < 4; ++kc) {
        v16h a = fragA16(h2 + (size_t)row * 128 + kc * 32, hi);
        v16h b = fragB16(wl + (size_t)m * 128 + kc * 32, hi);
        acc = __builtin_amdgcn_wmma_f32_16x16x32_f16(false, a, false, b, (short)0, acc, false, false);
      }
      if (m < 3) {
        float bb = b3[m];
        #pragma unroll
        for (int r = 0; r < 8; ++r) sres[(m0 + r + (hi ? 8 : 0)) * 3 + m] = acc[r] + bb;
      }
    }
  }
  __syncthreads();

  {                                              // finalize row tid
    int b = tid;
    float r0 = sres[b * 3 + 0], r1 = sres[b * 3 + 1], r2 = sres[b * 3 + 2];
    float dxp = scdx[b] + r0;
    float dyp = scdy[b] + r1;
    float ls = fminf(fmaxf(r2, -6.0f), 3.0f);
    const float* M = p2s + (size_t)b * 4;
    float sky0 = M[0] * dxp + M[1] * dyp;
    float sky1 = M[2] * dxp + M[3] * dyp;
    float* o = outp + (size_t)b * 54;
    o[0] = dxp; o[1] = dyp; o[2] = sky0; o[3] = sky1; o[4] = ls;
  }
}

// ---------------------------------------------------------------- launch

extern "C" void kernel_launch(void* const* d_in, const int* in_sizes, int n_in,
                              void* d_out, int out_size, void* d_ws, size_t ws_size,
                              hipStream_t stream) {
  (void)in_sizes; (void)n_in; (void)out_size; (void)ws_size;
  const float* rubin  = (const float*)d_in[0];
  const float* vis    = (const float*)d_in[1];
  const float* p2s    = (const float*)d_in[2];
  const float* rW0    = (const float*)d_in[3];
  const float* rWs    = (const float*)d_in[4];
  const float* rG     = (const float*)d_in[5];
  const float* rBt    = (const float*)d_in[6];
  const float* vW0    = (const float*)d_in[7];
  const float* vWs    = (const float*)d_in[8];
  const float* vG     = (const float*)d_in[9];
  const float* vBt    = (const float*)d_in[10];
  const float* projR  = (const float*)d_in[11];
  const float* projV  = (const float*)d_in[12];
  const float* logT   = (const float*)d_in[13];
  const float* W1     = (const float*)d_in[14];
  const float* b1     = (const float*)d_in[15];
  const float* W2     = (const float*)d_in[16];
  const float* b2     = (const float*)d_in[17];
  const float* W3     = (const float*)d_in[18];
  const float* b3     = (const float*)d_in[19];
  float* out = (float*)d_out;

  char* ws = (char*)d_ws;
  const size_t FM = (size_t)256 * 131072 * sizeof(_Float16);   // 64 MB per fmap buffer
  _Float16* X  = (_Float16*)(ws);
  _Float16* Y  = (_Float16*)(ws + FM);
  _Float16* Z  = (_Float16*)(ws + 2 * FM);
  float*    SW = (float*)(ws + 3 * FM);                        // 4 MB
  float*    RP = (float*)(ws + 3 * FM + 4194304);
  float*    VP = (float*)(ws + 3 * FM + 4194304 + 32768);
  float*    LG = (float*)(ws + 3 * FM + 4194304 + 65536);
  _Float16* PO = (_Float16*)(ws + 3 * FM + 4194304 + 131072);
  _Float16* H1 = (_Float16*)(ws + 3 * FM + 4194304 + 131072 + 65536);
  _Float16* H2 = (_Float16*)(ws + 3 * FM + 4194304 + 131072 + 131072);

  const int PACK_BLOCKS = (256 * 4096 * 32) / 256;

  // ---- rubin encoder: pack -> 4x (conv+GN+GELU) -> proj/l2n/sw (rw in Y, rp pooled)
  pack_nhwc32<<<PACK_BLOCKS, 256, 0, stream>>>(rubin, X, 6);
  conv_gn_gelu<<<256, 256, 0, stream>>>(X, Y, rW0,            6,  rG,      rBt);
  conv_gn_gelu<<<256, 256, 0, stream>>>(Y, X, rWs + 0 * 9216, 32, rG + 32, rBt + 32);
  conv_gn_gelu<<<256, 256, 0, stream>>>(X, Y, rWs + 1 * 9216, 32, rG + 64, rBt + 64);
  conv_gn_gelu<<<256, 256, 0, stream>>>(Y, X, rWs + 2 * 9216, 32, rG + 96, rBt + 96);
  proj_l2n<<<256, 256, 0, stream>>>(X, Y, projR, RP, SW, 1);

  // ---- vis encoder (reuses X; rubin rw stays in Y): vis_n in Z, vp pooled
  pack_nhwc32<<<PACK_BLOCKS, 256, 0, stream>>>(vis, X, 1);
  conv_gn_gelu<<<256, 256, 0, stream>>>(X, Z, vW0,            1,  vG,      vBt);
  conv_gn_gelu<<<256, 256, 0, stream>>>(Z, X, vWs + 0 * 9216, 32, vG + 32, vBt + 32);
  conv_gn_gelu<<<256, 256, 0, stream>>>(X, Z, vWs + 1 * 9216, 32, vG + 64, vBt + 64);
  conv_gn_gelu<<<256, 256, 0, stream>>>(Z, X, vWs + 2 * 9216, 32, vG + 96, vBt + 96);
  proj_l2n<<<256, 256, 0, stream>>>(X, Z, projV, VP, SW, 0);

  // ---- correlation logits (writes d_out cols 5..53)
  corr_logits<<<256, 256, 0, stream>>>(Y, Z, LG, out);

  // ---- softmax + pooled features + WMMA MLP + finalize (d_out cols 0..4)
  mlp_head<<<1, 256, 0, stream>>>(LG, RP, VP, logT, W1, b1, W2, b2, W3, b3, p2s,
                                  PO, H1, H2, out);
}
